// Block_36309653520993
// MI455X (gfx1250) — matrix operand order repaired and checked by reading, not compiled, not run on hardware
//
#include <hip/hip_runtime.h>

// ---------------------------------------------------------------------------
// Transformer block (B=2, T=2048, C=1024, H=16, D=64) for gfx1250 (MI455X).
// All matmuls run on V_WMMA_F32_16X16X32_BF16 (wave32), f32 accumulate.
// GEMM: 128x128 block tile, 8 waves (4x2), 32x64 per wave = 8 WMMAs/K-step,
// with double-buffered GLOBAL_LOAD_ASYNC_TO_LDS_B128 staging (ASYNCcnt).
// Attention: flash, async-DMA V staging overlapped with score WMMAs.
// ---------------------------------------------------------------------------

#define BB 2
#define TT 2048
#define CC 1024
#define HH 16
#define DD 64
#define MM (BB * TT)      // 4096 rows
#define C3 (3 * CC)       // 3072

typedef __attribute__((ext_vector_type(16))) __bf16 bf16x16;
typedef __attribute__((ext_vector_type(8)))  __bf16 bf16x8;
typedef __attribute__((ext_vector_type(8)))  float  f32x8;

static __device__ __forceinline__ f32x8 wmma_bf16(bf16x16 a, bf16x16 b, f32x8 c) {
  // D = A(16x32 bf16) x B(32x16 bf16) + C(16x16 f32)
  return __builtin_amdgcn_wmma_f32_16x16x32_bf16(false, a, false, b, (short)0, c,
                                                 false, false);
}

// Async DMA: global -> LDS, 16 bytes per lane, tracked by ASYNCcnt.
// VFLAT operand order per ISA 15.18: vdst = LDS-address VGPR, then 64-bit vaddr.
static __device__ __forceinline__ void async_copy_b128(unsigned int lds_off,
                                                       const void* gaddr) {
  asm volatile("global_load_async_to_lds_b128 %0, %1, off"
               :
               : "v"(lds_off), "v"((unsigned long long)(uintptr_t)gaddr)
               : "memory");
}
static __device__ __forceinline__ void wait_async_le4() {
  asm volatile("s_wait_asynccnt 0x4" ::: "memory");
}
static __device__ __forceinline__ void wait_async_0() {
  asm volatile("s_wait_asynccnt 0x0" ::: "memory");
}

// ---------------------------------------------------------------------------
// f32 -> bf16 conversion (weights)
// ---------------------------------------------------------------------------
__global__ void f2bf_kernel(const float* __restrict__ in, __bf16* __restrict__ out,
                            int n) {
  int i = blockIdx.x * blockDim.x + threadIdx.x;
  if (i < n) out[i] = (__bf16)in[i];
}

// ---------------------------------------------------------------------------
// LayerNorm (faithful: unbiased std, /(std+eps)), one 256-thread block per row.
// f32 in -> bf16 out (feeds WMMA GEMM).
// ---------------------------------------------------------------------------
__global__ __launch_bounds__(256) void layernorm_bf16_kernel(
    const float* __restrict__ x, const float* __restrict__ gamma,
    const float* __restrict__ beta, __bf16* __restrict__ out) {
  __shared__ float red[8];
  const int row = blockIdx.x, tid = threadIdx.x;
  const float* xr = x + (size_t)row * CC;
  float4 v = *(const float4*)&xr[tid * 4];
  float s = v.x + v.y + v.z + v.w;
#pragma unroll
  for (int off = 16; off > 0; off >>= 1) s += __shfl_xor(s, off, 32);
  if ((tid & 31) == 0) red[tid >> 5] = s;
  __syncthreads();
  float tot = 0.0f;
#pragma unroll
  for (int i = 0; i < 8; ++i) tot += red[i];
  const float mean = tot * (1.0f / CC);
  __syncthreads();
  const float d0 = v.x - mean, d1 = v.y - mean, d2 = v.z - mean, d3 = v.w - mean;
  float sq = d0 * d0 + d1 * d1 + d2 * d2 + d3 * d3;
#pragma unroll
  for (int off = 16; off > 0; off >>= 1) sq += __shfl_xor(sq, off, 32);
  if ((tid & 31) == 0) red[tid >> 5] = sq;
  __syncthreads();
  float sqt = 0.0f;
#pragma unroll
  for (int i = 0; i < 8; ++i) sqt += red[i];
  const float stdv = sqrtf(sqt * (1.0f / (CC - 1)));  // unbiased (Bessel)
  const float inv = 1.0f / (stdv + 1e-6f);
  const int c0 = tid * 4;
  __bf16* orow = out + (size_t)row * CC + c0;
  orow[0] = (__bf16)(gamma[c0 + 0] * d0 * inv + beta[c0 + 0]);
  orow[1] = (__bf16)(gamma[c0 + 1] * d1 * inv + beta[c0 + 1]);
  orow[2] = (__bf16)(gamma[c0 + 2] * d2 * inv + beta[c0 + 2]);
  orow[3] = (__bf16)(gamma[c0 + 3] * d3 * inv + beta[c0 + 3]);
}

// ---------------------------------------------------------------------------
// Generic bf16 WMMA GEMM: out[m,n] = sum_k A[m,k] * W[n,k]  (+bias,+relu,+resid)
// Block tile 128x128, 8 waves arranged 4(M) x 2(N); each wave computes 32x64:
// 2 A-fragments x 4 B-fragments -> 8 WMMAs per K-step of 32.
// Staging: double-buffered async DMA; last stage peeled so the steady-state
// loop is branch-free (issue next -> s_wait_asynccnt 4 -> barrier -> 8 WMMA).
// ---------------------------------------------------------------------------
__global__ __launch_bounds__(256) void gemm_bf16_kernel(
    const __bf16* __restrict__ A, const __bf16* __restrict__ Bw,
    const float* __restrict__ bias, const float* __restrict__ resid,
    float* __restrict__ outF, __bf16* __restrict__ outB,
    int M, int N, int K, int relu) {
  __shared__ __bf16 sA[2][128][32];
  __shared__ __bf16 sB[2][128][32];
  const int tid = threadIdx.x;
  const int wid = tid >> 5, lane = tid & 31;
  const int wr = wid >> 1, wc = wid & 1;
  const int l15 = lane & 15, grp = lane >> 4;
  const int m0 = blockIdx.y * 128, n0 = blockIdx.x * 128;
  const int srow = tid >> 1;          // 0..127 staged row
  const int scol = (tid & 1) * 16;    // 0 or 16 (elem offset)

  const __bf16* gA = &A[(size_t)(m0 + srow) * K + scol];
  const __bf16* gB = &Bw[(size_t)(n0 + srow) * K + scol];

  // LDS destination addresses, fixed per buffer parity (hoisted out of loop)
  unsigned ldsA[2], ldsB[2];
  ldsA[0] = (unsigned)(uintptr_t)&sA[0][srow][scol];
  ldsA[1] = (unsigned)(uintptr_t)&sA[1][srow][scol];
  ldsB[0] = (unsigned)(uintptr_t)&sB[0][srow][scol];
  ldsB[1] = (unsigned)(uintptr_t)&sB[1][srow][scol];

  // per-thread async stage: 32B of A-tile + 32B of B-tile (4 x b128 DMA ops)
  auto issue = [&](int st) {
    const int buf = st & 1;
    const __bf16* a = gA + st * 32;
    const __bf16* b = gB + st * 32;
    async_copy_b128(ldsA[buf],      a);
    async_copy_b128(ldsA[buf] + 16, a + 8);
    async_copy_b128(ldsB[buf],      b);
    async_copy_b128(ldsB[buf] + 16, b + 8);
  };

  f32x8 acc[2][4];
#pragma unroll
  for (int ms = 0; ms < 2; ++ms)
#pragma unroll
    for (int nt = 0; nt < 4; ++nt) acc[ms][nt] = (f32x8){};

  auto compute = [&](int buf) {
    // A fragments (16x32, ISA 7.12.2 16-bit A layout):
    //   lane: row m = l15, v[j]=A[m][grp*8+j], v[8+j]=A[m][grp*8+16+j]
    bf16x16 af[2];
#pragma unroll
    for (int ms = 0; ms < 2; ++ms) {
      const __bf16* p = &sA[buf][wr * 32 + ms * 16 + l15][grp * 8];
      bf16x8 lo = *(const bf16x8*)p, hi = *(const bf16x8*)(p + 16);
#pragma unroll
      for (int j = 0; j < 8; ++j) { af[ms][j] = lo[j]; af[ms][8 + j] = hi[j]; }
    }
    // B fragments (32x16): lane: col n = l15, v[j] = Bmat[grp*8+j][n], v[8+j] = Bmat[grp*8+16+j][n] = W[n][k..]
    bf16x16 bfr[4];
#pragma unroll
    for (int nt = 0; nt < 4; ++nt) {
      const __bf16* p = &sB[buf][wc * 64 + nt * 16 + l15][grp * 8];
      bf16x8 lo = *(const bf16x8*)p, hi = *(const bf16x8*)(p + 16);
#pragma unroll
      for (int j = 0; j < 8; ++j) { bfr[nt][j] = lo[j]; bfr[nt][8 + j] = hi[j]; }
    }
#pragma unroll
    for (int ms = 0; ms < 2; ++ms)
#pragma unroll
      for (int nt = 0; nt < 4; ++nt)
        acc[ms][nt] = wmma_bf16(af[ms], bfr[nt], acc[ms][nt]);
  };

  const int nst = K / 32;  // >= 2 for all our shapes
  issue(0);
  for (int st = 0; st < nst - 1; ++st) {
    issue(st + 1);    // buffer (st+1)&1 free: stage st-1 consumed (last barrier)
    wait_async_le4(); // in-order ASYNCcnt: stage st fully landed in LDS
    __syncthreads();
    compute(st & 1);
    __syncthreads();  // stage st consumed by all waves -> its buffer reusable
  }
  wait_async_0();
  __syncthreads();
  compute((nst - 1) & 1);

  // epilogue: C/D layout -> row m = i + grp*8, col n = l15
#pragma unroll
  for (int ms = 0; ms < 2; ++ms) {
#pragma unroll
    for (int nt = 0; nt < 4; ++nt) {
#pragma unroll
      for (int i = 0; i < 8; ++i) {
        const int m = m0 + wr * 32 + ms * 16 + grp * 8 + i;
        const int n = n0 + wc * 64 + nt * 16 + l15;
        float v = acc[ms][nt][i] + (bias ? bias[n] : 0.0f);
        if (relu) v = fmaxf(v, 0.0f);
        if (resid) v += resid[(size_t)m * N + n];
        if (outF) outF[(size_t)m * N + n] = v;
        if (outB) outB[(size_t)m * N + n] = (__bf16)v;
      }
    }
  }
}

// ---------------------------------------------------------------------------
// Causal flash attention. qkv: bf16 [B*T, 3C]; out: bf16 [B*T, C].
// One wave per 16-query tile; 8 waves/block. Streams 32 keys per chunk:
//   async V DMA issued first (lands during score phase),
//   S^T (keys x queries) via 4 WMMAs  -> lane-local softmax stats (+1 shfl),
//   s_wait_asynccnt 0, then O^T (dvals x queries) via 4 WMMAs with per-lane
//   flash rescale (alpha and 1/l uniform per lane).
// ---------------------------------------------------------------------------
__global__ __launch_bounds__(256) void attn_kernel(const __bf16* __restrict__ qkv,
                                                   __bf16* __restrict__ outb) {
  __shared__ __bf16 vbuf[8][32][DD];  // per-wave V chunk (32 keys x 64 d)
  __shared__ __bf16 pbuf[8][32][16];  // per-wave P^T (32 keys x 16 queries)
  const int tid = threadIdx.x, wid = tid >> 5, lane = tid & 31;
  const int l15 = lane & 15, grp = lane >> 4;
  const int tileId = blockIdx.x * 8 + wid;
  const int tilesPerBH = TT / 16;
  const int bh = tileId / tilesPerBH;
  const int t0 = (tileId % tilesPerBH) * 16;
  const int bi = bh / HH, h = bh % HH;
  const __bf16* qp = qkv + (size_t)bi * TT * C3 + h * DD;
  const __bf16* kp = qp + CC;
  const __bf16* vp = qp + 2 * CC;
  const unsigned vdst = (unsigned)(uintptr_t)&vbuf[wid][lane][0];

  // Q as WMMA-B fragments (B[d][t] = Q[t][d]); lane: col t = l15
  bf16x16 qb0, qb1;
  {
    const __bf16* r = qp + (size_t)(t0 + l15) * C3 + grp * 8;
    bf16x8 a0 = *(const bf16x8*)r, a1 = *(const bf16x8*)(r + 16);
#pragma unroll
    for (int j = 0; j < 8; ++j) { qb0[j] = a0[j]; qb0[8 + j] = a1[j]; }
    a0 = *(const bf16x8*)(r + 32); a1 = *(const bf16x8*)(r + 48);
#pragma unroll
    for (int j = 0; j < 8; ++j) { qb1[j] = a0[j]; qb1[8 + j] = a1[j]; }
  }

  f32x8 oacc[4] = {{}, {}, {}, {}};   // O^T tiles: M=d (16 each), N=t
  float m_run = -3.0e30f, l_run = 0.0f;
  const int t_lane = t0 + l15;
  const int t_hi = t0 + 15;

  for (int s0 = 0; s0 <= t_hi; s0 += 32) {
    // ---- kick off async V DMA for this chunk (overlaps with score phase) --
    {
      int s = s0 + lane; if (s > TT - 1) s = TT - 1;
      const __bf16* r = vp + (size_t)s * C3;
#pragma unroll
      for (int j = 0; j < 8; ++j)
        async_copy_b128(vdst + j * 16, r + j * 8);
    }

    // ---- scores S^T = K * Q^T for two 16-key subtiles --------------------
    f32x8 sc0 = {}, sc1 = {};
    {
      int s = s0 + l15; if (s > TT - 1) s = TT - 1;
      const __bf16* r = kp + (size_t)s * C3 + grp * 8;
      bf16x16 ka, kb2;
      bf16x8 x0 = *(const bf16x8*)r, x1 = *(const bf16x8*)(r + 16);
#pragma unroll
      for (int j = 0; j < 8; ++j) { ka[j] = x0[j]; ka[8 + j] = x1[j]; }
      x0 = *(const bf16x8*)(r + 32); x1 = *(const bf16x8*)(r + 48);
#pragma unroll
      for (int j = 0; j < 8; ++j) { kb2[j] = x0[j]; kb2[8 + j] = x1[j]; }
      sc0 = wmma_bf16(ka, qb0, sc0);   // d = 0..31
      sc0 = wmma_bf16(kb2, qb1, sc0);  // d = 32..63
    }
    {
      int s = s0 + 16 + l15; if (s > TT - 1) s = TT - 1;
      const __bf16* r = kp + (size_t)s * C3 + grp * 8;
      bf16x16 ka, kb2;
      bf16x8 x0 = *(const bf16x8*)r, x1 = *(const bf16x8*)(r + 16);
#pragma unroll
      for (int j = 0; j < 8; ++j) { ka[j] = x0[j]; ka[8 + j] = x1[j]; }
      x0 = *(const bf16x8*)(r + 32); x1 = *(const bf16x8*)(r + 48);
#pragma unroll
      for (int j = 0; j < 8; ++j) { kb2[j] = x0[j]; kb2[8 + j] = x1[j]; }
      sc1 = wmma_bf16(ka, qb0, sc1);
      sc1 = wmma_bf16(kb2, qb1, sc1);
    }

    // ---- scale + causal mask; online softmax stats -----------------------
    float mx = -3.0e30f;
#pragma unroll
    for (int i = 0; i < 8; ++i) {
      const int sa = s0 + grp * 8 + i;
      sc0[i] = (sa <= t_lane) ? sc0[i] * 0.125f : -3.0e30f;  // D^-0.5 = 1/8
      mx = fmaxf(mx, sc0[i]);
      const int sb = s0 + 16 + grp * 8 + i;
      sc1[i] = (sb <= t_lane) ? sc1[i] * 0.125f : -3.0e30f;
      mx = fmaxf(mx, sc1[i]);
    }
    mx = fmaxf(mx, __shfl_xor(mx, 16, 32));
    const float m_new = fmaxf(m_run, mx);
    const float alpha = __expf(m_run - m_new);
    float csum = 0.0f;
#pragma unroll
    for (int i = 0; i < 8; ++i) {
      const float p0 = __expf(sc0[i] - m_new);
      const float p1 = __expf(sc1[i] - m_new);
      csum += p0 + p1;
      pbuf[wid][grp * 8 + i][l15]      = (__bf16)p0;   // P^T[s_local][t]
      pbuf[wid][16 + grp * 8 + i][l15] = (__bf16)p1;
    }
    csum += __shfl_xor(csum, 16, 32);
    l_run = l_run * alpha + csum;
    m_run = m_new;

    // ---- V chunk must be resident now ------------------------------------
    wait_async_0();

    // ---- O^T += V^T * P^T  (A = V^T fragments, B = P^T fragments) --------
    bf16x16 pb;
#pragma unroll
    for (int j = 0; j < 16; ++j) pb[j] = pbuf[wid][(j < 8 ? grp * 8 + j : 8 + grp * 8 + j)][l15];
#pragma unroll
    for (int dt = 0; dt < 4; ++dt) {
      bf16x16 va;
#pragma unroll
      for (int j = 0; j < 8; ++j) {
        va[j]     = vbuf[wid][grp * 8 + j][dt * 16 + l15];
        va[8 + j] = vbuf[wid][grp * 8 + 16 + j][dt * 16 + l15];
      }
#pragma unroll
      for (int i = 0; i < 8; ++i) oacc[dt][i] *= alpha;
      oacc[dt] = wmma_bf16(va, pb, oacc[dt]);
    }
  }

  // ---- finalize: out[t][d] = O^T[d][t] / l -------------------------------
  const float inv_l = 1.0f / l_run;
  __bf16* orow = outb + (size_t)(bi * TT + t0 + l15) * CC + h * DD;
#pragma unroll
  for (int dt = 0; dt < 4; ++dt) {
    bf16x8 v8;
#pragma unroll
    for (int i = 0; i < 8; ++i) v8[i] = (__bf16)(oacc[dt][i] * inv_l);
    *(bf16x8*)&orow[dt * 16 + grp * 8] = v8;  // d = dt*16 + grp*8 + i
  }
}

// ---------------------------------------------------------------------------
// Launcher
// ---------------------------------------------------------------------------
extern "C" void kernel_launch(void* const* d_in, const int* in_sizes, int n_in,
                              void* d_out, int out_size, void* d_ws, size_t ws_size,
                              hipStream_t stream) {
  (void)in_sizes; (void)n_in; (void)out_size; (void)ws_size;
  const float* x      = (const float*)d_in[0];
  const float* qkv_w  = (const float*)d_in[1];
  const float* proj_w = (const float*)d_in[2];
  const float* proj_b = (const float*)d_in[3];
  const float* l1_w   = (const float*)d_in[4];
  const float* l1_b   = (const float*)d_in[5];
  const float* l3_w   = (const float*)d_in[6];
  const float* l3_b   = (const float*)d_in[7];
  const float* ln1_g  = (const float*)d_in[8];
  const float* ln1_bv = (const float*)d_in[9];
  const float* ln2_g  = (const float*)d_in[10];
  const float* ln2_bv = (const float*)d_in[11];
  float* out = (float*)d_out;

  // workspace carve-up (256B aligned)
  char* p = (char*)d_ws;
  auto take = [&](size_t bytes) -> char* {
    char* r = p;
    p += (bytes + 255) & ~(size_t)255;
    return r;
  };
  __bf16* wqkv  = (__bf16*)take((size_t)3 * CC * CC * 2);
  __bf16* wproj = (__bf16*)take((size_t)CC * CC * 2);
  __bf16* wl1   = (__bf16*)take((size_t)4 * CC * CC * 2);
  __bf16* wl3   = (__bf16*)take((size_t)4 * CC * CC * 2);
  __bf16* ln1b  = (__bf16*)take((size_t)MM * CC * 2);
  __bf16* qkvb  = (__bf16*)take((size_t)MM * C3 * 2);
  __bf16* attnb = (__bf16*)take((size_t)MM * CC * 2);
  float*  x1    = (float*)take((size_t)MM * CC * 4);
  __bf16* ln2b  = (__bf16*)take((size_t)MM * CC * 2);
  __bf16* hbuf  = (__bf16*)take((size_t)MM * 4 * CC * 2);

  // 1) weights -> bf16
  {
    int n;
    n = 3 * CC * CC; f2bf_kernel<<<(n + 255) / 256, 256, 0, stream>>>(qkv_w, wqkv, n);
    n = CC * CC;     f2bf_kernel<<<(n + 255) / 256, 256, 0, stream>>>(proj_w, wproj, n);
    n = 4 * CC * CC; f2bf_kernel<<<(n + 255) / 256, 256, 0, stream>>>(l1_w, wl1, n);
    n = 4 * CC * CC; f2bf_kernel<<<(n + 255) / 256, 256, 0, stream>>>(l3_w, wl3, n);
  }
  // 2) ln1(x) -> bf16
  layernorm_bf16_kernel<<<MM, 256, 0, stream>>>(x, ln1_g, ln1_bv, ln1b);
  // 3) qkv = ln1 @ qkv_w^T   [4096 x 3072]
  gemm_bf16_kernel<<<dim3(C3 / 128, MM / 128), 256, 0, stream>>>(
      ln1b, wqkv, nullptr, nullptr, nullptr, qkvb, MM, C3, CC, 0);
  // 4) causal attention -> bf16 [4096 x 1024]
  attn_kernel<<<(BB * HH * (TT / 16)) / 8, 256, 0, stream>>>(qkvb, attnb);
  // 5) x1 = x + attn @ proj_w^T + proj_b  (f32)
  gemm_bf16_kernel<<<dim3(CC / 128, MM / 128), 256, 0, stream>>>(
      attnb, wproj, proj_b, x, x1, nullptr, MM, CC, CC, 0);
  // 6) ln2(x1) -> bf16
  layernorm_bf16_kernel<<<MM, 256, 0, stream>>>(x1, ln2_g, ln2_bv, ln2b);
  // 7) h = relu(ln2 @ l1_w^T + l1_b) -> bf16 [4096 x 4096]
  gemm_bf16_kernel<<<dim3(4 * CC / 128, MM / 128), 256, 0, stream>>>(
      ln2b, wl1, l1_b, nullptr, nullptr, hbuf, MM, 4 * CC, CC, 1);
  // 8) out = x1 + h @ l3_w^T + l3_b  (f32)
  gemm_bf16_kernel<<<dim3(CC / 128, MM / 128), 256, 0, stream>>>(
      hbuf, wl3, l3_b, x1, out, nullptr, MM, CC, 4 * CC, 0);
}
